// MyDeformableAttention_74990128988476
// MI455X (gfx1250) — matrix operand order repaired and checked by reading, not compiled, not run on hardware
//
#include <hip/hip_runtime.h>
#include <hip/hip_bf16.h>
#include <math.h>

// ---------------------------------------------------------------------------
// MI455X (gfx1250) deformable attention.
// Wave32, WMMA bf16 (16x16x32) for all GEMMs + flash-attention core.
// This round: GEMM tiles staged in LDS via GLOBAL_LOAD_ASYNC_TO_LDS_B128
// with ASYNCcnt double-buffering (fetch of tile i+1 overlaps WMMA of tile i).
// Pipeline:
//   1) q = features * q_w + q_b                      (WMMA GEMM, async-LDS)
//   2) offsets: dw3x3 s2 + LN + GELU + pw(2) + tanh, bilinear sample -> xs
//   3) k = xs * k_w + k_b ; v = xs * v_w + v_b       (WMMA GEMM, async-LDS)
//   4) flash attention per (head, 16-row m-tile)     (WMMA, online softmax)
//   5) out = attn_out * o_w + o_b                    (WMMA GEMM) -> d_out
// ---------------------------------------------------------------------------

typedef __attribute__((ext_vector_type(16))) __bf16 v16bf;
typedef __attribute__((ext_vector_type(8)))  float  v8f;

union ABf {
    v16bf v;
    unsigned short u[16];
};

// fp32 -> bf16 bits, round to nearest even
static __device__ __forceinline__ unsigned short f2bf(float f) {
    unsigned int x = __float_as_uint(f);
    x += 0x7FFFu + ((x >> 16) & 1u);
    return (unsigned short)(x >> 16);
}

static __device__ __forceinline__ v8f wmma_bf16(v16bf a, v16bf b, v8f c) {
    // D = A(16x32 bf16) x B(32x16 bf16) + C(16x16 f32)
    return __builtin_amdgcn_wmma_f32_16x16x32_bf16(
        /*neg_a=*/false, a, /*neg_b=*/false, b,
        /*c_mod=*/(short)0, c, /*reuse_a=*/false, /*reuse_b=*/false);
}

// A-fragment K index for 16-bit 16x32 layout (pair base for VGPR v, wave half)
static __device__ __forceinline__ int a_kidx(int v, int half) {
    return ((v >> 2) << 4) + ((v & 3) << 1) + (half << 3);
}
// B-fragment K index for 16-bit 32x16 layout
static __device__ __forceinline__ int b_kidx(int v, int half) {
    return ((v >> 2) << 4) + ((v & 3) << 1) + (half << 3);
}

// ---- gfx1250 async global->LDS copy (ASYNCcnt path), inline asm -----------
static __device__ __forceinline__ unsigned lds_off(const void* p) {
    // low 32 bits of a generic pointer to LDS = LDS byte address
    return (unsigned)(unsigned long long)p;
}
static __device__ __forceinline__ void async_ld_b128(unsigned lds, const float* g) {
    asm volatile("global_load_async_to_lds_b128 %0, %1, off"
                 :: "v"(lds), "v"((unsigned long long)g)
                 : "memory");
}
static __device__ __forceinline__ void wait_async_le(int prevInFlight) {
    if (prevInFlight) {
        asm volatile("s_wait_asynccnt 0x8" ::: "memory");
    } else {
        asm volatile("s_wait_asynccnt 0x0" ::: "memory");
    }
}

// ---------------------------------------------------------------------------
// Generic channel-major GEMM:
//   out[b, n, m] = sum_k in[b, k, m] * w[n, k] + bias[n]
// One wave computes a 16(m) x 16(n) tile; K-loop in steps of 32 (bf16 WMMA).
// A tile (32k x 16m fp32) and W tile (16n x 32k fp32) are staged in LDS by
// GLOBAL_LOAD_ASYNC_TO_LDS_B128 (8 instrs / tile), double-buffered on
// ASYNCcnt so the fetch of tile i+1 overlaps conversion+WMMA of tile i.
// ---------------------------------------------------------------------------
#define GEMM_WAVES 4
#define TILE_F 1024  // floats per staged tile pair: 512 (A) + 512 (W)

__global__ void gemm_cm_bf16(const float* __restrict__ in,
                             const float* __restrict__ w,
                             const float* __restrict__ bias,
                             float* __restrict__ out,
                             int Bn, int M, int K, int N) {
    __shared__ float stage[GEMM_WAVES][2][TILE_F];  // 32 KB / block

    const int lane = threadIdx.x & 31;
    const int wave = threadIdx.x >> 5;
    const int wavesPerBlock = blockDim.x >> 5;
    const int mTiles = M >> 4;
    const int nTiles = N >> 4;
    const long long totalTiles = (long long)Bn * mTiles * nTiles;
    long long tile = (long long)blockIdx.x * wavesPerBlock + wave;
    if (tile >= totalTiles) return;  // uniform per wave -> EXEC all-ones inside

    const int b   = (int)(tile / (mTiles * nTiles));
    const int rem = (int)(tile % (mTiles * nTiles));
    const int mt  = rem % mTiles;
    const int nt  = rem / mTiles;

    const int half = lane >> 4;
    const int ln   = lane & 15;
    const int n    = nt * 16 + ln;          // B-frag / D col for this lane

    const float* __restrict__ Ab = in + (long long)b * K * M + mt * 16;
    const float* __restrict__ Wb = w + (long long)(nt * 16) * K;

    // ---- async tile issue: A rows k-major (32 x 16f), W rows n-major (16 x 32f)
    auto issue_tile = [&](int k0, int bb) {
        float* As = &stage[wave][bb][0];
        float* Ws = &stage[wave][bb][512];
        const unsigned abase = lds_off(As);
        const unsigned wbase = lds_off(Ws);
        // A: lane = k row, 4 x 16B segments along m
        const float* gA = Ab + (long long)(k0 + lane) * M;
#pragma unroll
        for (int i = 0; i < 4; ++i)
            async_ld_b128(abase + lane * 64 + i * 16, gA + i * 4);
        // W: 16 rows x 128B; lane covers (n = lane&15, seg = i*2 + lane>>4)
        const float* gW = Wb + (long long)(lane & 15) * K + k0;
        const unsigned wrow = wbase + (unsigned)(lane & 15) * 128;
#pragma unroll
        for (int i = 0; i < 4; ++i) {
            const int s = i * 2 + (lane >> 4);
            async_ld_b128(wrow + s * 16, gW + s * 4);
        }
    };

    issue_tile(0, 0);
    v8f acc = {};
    int bb = 0;
    for (int k0 = 0; k0 < K; k0 += 32) {
        const int hasNext = (k0 + 32 < K);
        if (hasNext) issue_tile(k0 + 32, bb ^ 1);
        wait_async_le(hasNext);  // tile bb resident; next tile still in flight

        const float* As = &stage[wave][bb][0];
        const float* Ws = &stage[wave][bb][512];
        ABf a, bf;
#pragma unroll
        for (int vr = 0; vr < 8; ++vr) {
            const int ka = a_kidx(vr, half);
            a.u[2 * vr]     = f2bf(As[ka * 16 + ln]);
            a.u[2 * vr + 1] = f2bf(As[(ka + 1) * 16 + ln]);
            const int kb = b_kidx(vr, half);
            bf.u[2 * vr]     = f2bf(Ws[ln * 32 + kb]);
            bf.u[2 * vr + 1] = f2bf(Ws[ln * 32 + kb + 1]);
        }
        acc = wmma_bf16(a.v, bf.v, acc);
        bb ^= 1;
    }

    const float bn = bias ? bias[n] : 0.0f;
    // D layout: row = r + 8*half, col = ln  ->  contiguous m per lane
    float* __restrict__ Ob = out + ((long long)b * N + n) * M + mt * 16 + 8 * half;
#pragma unroll
    for (int r = 0; r < 8; ++r) Ob[r] = acc[r] + bn;
}

// ---------------------------------------------------------------------------
// Offset branch + bilinear sampling (fused, elementwise; VALU-bound, tiny).
// One thread per (bg, oy, ox). Produces xs[bg*64 + c][p] channel-major.
// ---------------------------------------------------------------------------
__global__ void offset_sample(const float* __restrict__ q,        // (16,64,64,64)
                              const float* __restrict__ features, // (16,64,64,64)
                              const float* __restrict__ dw_w,     // (64,1,3,3)
                              const float* __restrict__ dw_b,     // (64)
                              const float* __restrict__ ln_g,
                              const float* __restrict__ ln_b,
                              const float* __restrict__ pw_w,     // (2,64)
                              float* __restrict__ xs)             // (16,64,1024)
{
    const int HK = 32, WK = 32;
    int idx = blockIdx.x * blockDim.x + threadIdx.x;
    if (idx >= 16 * HK * WK) return;
    const int bg = idx / (HK * WK);
    const int p  = idx % (HK * WK);
    const int oy = p / WK, ox = p % WK;

    const float* __restrict__ qb = q + (long long)bg * 64 * 64 * 64;

    // depthwise 3x3 stride-2 pad-1 over 64 group channels
    float xc[64];
    float mu = 0.0f;
#pragma unroll 4
    for (int c = 0; c < 64; ++c) {
        float s = dw_b[c];
#pragma unroll
        for (int ky = 0; ky < 3; ++ky) {
            const int iy = oy * 2 + ky - 1;
            if (iy < 0 || iy >= 64) continue;
#pragma unroll
            for (int kx = 0; kx < 3; ++kx) {
                const int ix = ox * 2 + kx - 1;
                if (ix < 0 || ix >= 64) continue;
                s += qb[((long long)c * 64 + iy) * 64 + ix] * dw_w[c * 9 + ky * 3 + kx];
            }
        }
        xc[c] = s;
        mu += s;
    }
    mu *= (1.0f / 64.0f);
    float var = 0.0f;
#pragma unroll 4
    for (int c = 0; c < 64; ++c) { float d = xc[c] - mu; var += d * d; }
    var *= (1.0f / 64.0f);
    const float rstd = rsqrtf(var + 1e-5f);

    // LN + exact GELU + pointwise (2 ch)
    float o0 = 0.0f, o1 = 0.0f;
#pragma unroll 4
    for (int c = 0; c < 64; ++c) {
        float t = (xc[c] - mu) * rstd * ln_g[c] + ln_b[c];
        t = 0.5f * t * (1.0f + erff(t * 0.70710678118654752f));
        o0 += t * pw_w[c];        // offset ch0 -> y
        o1 += t * pw_w[64 + c];   // offset ch1 -> x
    }
    const float offy = tanhf(o0) * (2.0f / 32.0f);
    const float offx = tanhf(o1) * (2.0f / 32.0f);
    const float refy = ((float)oy + 0.5f) * (2.0f / 31.0f) - 1.0f;
    const float refx = ((float)ox + 0.5f) * (2.0f / 31.0f) - 1.0f;
    const float py = refy + offy;
    const float px = refx + offx;

    // grid[...,0] = x, grid[...,1] = y ; map [-1,1] -> [0,63]
    const float gx = (px + 1.0f) * 0.5f * 63.0f;
    const float gy = (py + 1.0f) * 0.5f * 63.0f;
    const float x0f = floorf(gx), y0f = floorf(gy);
    const int   x0 = (int)x0f, y0 = (int)y0f;
    const float wx1 = gx - x0f, wy1 = gy - y0f;

    const int vx0 = (x0 >= 0 && x0 <= 63), vx1 = (x0 + 1 >= 0 && x0 + 1 <= 63);
    const int vy0 = (y0 >= 0 && y0 <= 63), vy1 = (y0 + 1 >= 0 && y0 + 1 <= 63);
    const int cx0 = min(max(x0, 0), 63),     cx1 = min(max(x0 + 1, 0), 63);
    const int cy0 = min(max(y0, 0), 63),     cy1 = min(max(y0 + 1, 0), 63);

    const float w00 = (1.0f - wx1) * (1.0f - wy1) * (float)(vx0 & vy0);
    const float w10 = wx1 * (1.0f - wy1) * (float)(vx1 & vy0);
    const float w01 = (1.0f - wx1) * wy1 * (float)(vx0 & vy1);
    const float w11 = wx1 * wy1 * (float)(vx1 & vy1);

    const int i00 = cy0 * 64 + cx0, i10 = cy0 * 64 + cx1;
    const int i01 = cy1 * 64 + cx0, i11 = cy1 * 64 + cx1;

    const float* __restrict__ fb = features + (long long)bg * 64 * 4096;
    float* __restrict__ xo = xs + (long long)bg * 64 * 1024 + p;
#pragma unroll 4
    for (int c = 0; c < 64; ++c) {
        const float* fc = fb + (long long)c * 4096;
        xo[(long long)c * 1024] =
            w00 * fc[i00] + w10 * fc[i10] + w01 * fc[i01] + w11 * fc[i11];
    }
}

// ---------------------------------------------------------------------------
// Flash attention core.  One wave per (bh, m-tile of 16 query pixels).
//   q : (32, 32, 4096)  channel-major (bh, c, m)
//   k : (32, 32, 1024)  channel-major (bh, c, n)
//   v : (32, 32, 1024)
//   o : (32, 32, 4096)
// scores tile = 1 WMMA (K = HC = 32); online softmax; P.V = 2 WMMAs per
// 32-wide n-block via a 1 KB LDS C->A layout transpose.
// ---------------------------------------------------------------------------
__global__ void flash_attn_wmma(const float* __restrict__ q,
                                const float* __restrict__ kbuf,
                                const float* __restrict__ vbuf,
                                float* __restrict__ obuf) {
    __shared__ unsigned short Pld[4][16 * 32];  // per-wave P staging (bf16 bits)

    const int lane = threadIdx.x & 31;
    const int wv   = threadIdx.x >> 5;
    const long long wid = (long long)blockIdx.x * 4 + wv;
    const int MT = 4096 / 16;  // 256 m-tiles per head-batch
    if (wid >= 32LL * MT) return;
    const int bh = (int)(wid / MT);
    const int mt = (int)(wid % MT);

    const int half = lane >> 4;
    const int ln   = lane & 15;
    const float scale = 0.17677669529663689f;  // 1/sqrt(HC=32)

    // q A-fragment (held across the whole n loop), scale folded in
    ABf qa;
    {
        const int m = mt * 16 + ln;
#pragma unroll
        for (int vr = 0; vr < 8; ++vr) {
            const int c = a_kidx(vr, half);
            qa.u[2 * vr]     = f2bf(q[((long long)bh * 32 + c)     * 4096 + m] * scale);
            qa.u[2 * vr + 1] = f2bf(q[((long long)bh * 32 + c + 1) * 4096 + m] * scale);
        }
    }

    v8f o0 = {}, o1 = {};
    float rmax[8], rsum[8];
#pragma unroll
    for (int r = 0; r < 8; ++r) { rmax[r] = -INFINITY; rsum[r] = 0.0f; }

    unsigned short* __restrict__ Pw = &Pld[wv][0];

    for (int n0 = 0; n0 < 1024; n0 += 32) {
        // hint next k/v slabs into cache while we compute this block
        if (n0 + 32 < 1024) {
            __builtin_prefetch(kbuf + ((long long)bh * 32 + ln) * 1024 + n0 + 32, 0, 1);
            __builtin_prefetch(vbuf + ((long long)bh * 32 + ln) * 1024 + n0 + 32, 0, 1);
        }
        // ---- scores: two 16x16 tiles, K = 32 channels, single WMMA each ----
        ABf kb0, kb1;
#pragma unroll
        for (int vr = 0; vr < 8; ++vr) {
            const int c = b_kidx(vr, half);
            const long long r0 = ((long long)bh * 32 + c) * 1024;
            const long long r1 = ((long long)bh * 32 + c + 1) * 1024;
            kb0.u[2 * vr]     = f2bf(kbuf[r0 + n0 + ln]);
            kb0.u[2 * vr + 1] = f2bf(kbuf[r1 + n0 + ln]);
            kb1.u[2 * vr]     = f2bf(kbuf[r0 + n0 + 16 + ln]);
            kb1.u[2 * vr + 1] = f2bf(kbuf[r1 + n0 + 16 + ln]);
        }
        v8f z = {};
        v8f s0 = wmma_bf16(qa.v, kb0.v, z);
        v8f s1 = wmma_bf16(qa.v, kb1.v, z);

        // ---- online softmax: per-row max/sum across 16 lanes of each half ----
        float al[8];
#pragma unroll
        for (int r = 0; r < 8; ++r) {
            float t = fmaxf(s0[r], s1[r]);
#pragma unroll
            for (int msk = 1; msk < 16; msk <<= 1)
                t = fmaxf(t, __shfl_xor(t, msk, 32));
            const float nmax = fmaxf(rmax[r], t);
            al[r] = __expf(rmax[r] - nmax);
            rmax[r] = nmax;
        }
#pragma unroll
        for (int r = 0; r < 8; ++r) {
            const float e0 = __expf(s0[r] - rmax[r]);
            const float e1 = __expf(s1[r] - rmax[r]);
            const int row = r + 8 * half;          // C-layout row for this lane
            Pw[row * 32 + ln]      = f2bf(e0);
            Pw[row * 32 + 16 + ln] = f2bf(e1);
            float rs = e0 + e1;
#pragma unroll
            for (int msk = 1; msk < 16; msk <<= 1)
                rs += __shfl_xor(rs, msk, 32);
            rsum[r] = rsum[r] * al[r] + rs;
            o0[r] *= al[r];
            o1[r] *= al[r];
        }

        // ---- P.V: read P back in A layout (same-wave LDS, in-order DS) ----
        ABf pa;
#pragma unroll
        for (int vr = 0; vr < 8; ++vr) {
            const int kk = a_kidx(vr, half);
            pa.u[2 * vr]     = Pw[ln * 32 + kk];
            pa.u[2 * vr + 1] = Pw[ln * 32 + kk + 1];
        }
        ABf vb0, vb1;
#pragma unroll
        for (int vr = 0; vr < 8; ++vr) {
            const int kk = b_kidx(vr, half);  // local n index — contiguous loads
            const float* vr0 = vbuf + ((long long)bh * 32 + ln)      * 1024 + n0 + kk;
            const float* vr1 = vbuf + ((long long)bh * 32 + 16 + ln) * 1024 + n0 + kk;
            vb0.u[2 * vr]     = f2bf(vr0[0]);
            vb0.u[2 * vr + 1] = f2bf(vr0[1]);
            vb1.u[2 * vr]     = f2bf(vr1[0]);
            vb1.u[2 * vr + 1] = f2bf(vr1[1]);
        }
        o0 = wmma_bf16(pa.v, vb0.v, o0);
        o1 = wmma_bf16(pa.v, vb1.v, o1);
    }

    // ---- finalize: divide by row sums, store channel-major ----
#pragma unroll
    for (int r = 0; r < 8; ++r) {
        const int m = mt * 16 + r + 8 * half;
        const float inv = 1.0f / rsum[r];
        obuf[((long long)bh * 32 + ln)      * 4096 + m] = o0[r] * inv;
        obuf[((long long)bh * 32 + 16 + ln) * 4096 + m] = o1[r] * inv;
    }
}

// ---------------------------------------------------------------------------
// Host-side orchestration
// ---------------------------------------------------------------------------
extern "C" void kernel_launch(void* const* d_in, const int* in_sizes, int n_in,
                              void* d_out, int out_size, void* d_ws, size_t ws_size,
                              hipStream_t stream) {
    (void)in_sizes; (void)n_in; (void)out_size;

    const float* features = (const float*)d_in[1];   // (4,256,64,64)
    const float* off_dw_w = (const float*)d_in[2];   // (64,1,3,3)
    const float* off_dw_b = (const float*)d_in[3];
    const float* ln_g     = (const float*)d_in[4];
    const float* ln_b     = (const float*)d_in[5];
    const float* off_pw_w = (const float*)d_in[6];   // (2,64)
    const float* q_w      = (const float*)d_in[7];   // (256,256)
    const float* q_b      = (const float*)d_in[8];
    const float* k_w      = (const float*)d_in[9];
    const float* k_b      = (const float*)d_in[10];
    const float* v_w      = (const float*)d_in[11];
    const float* v_b      = (const float*)d_in[12];
    const float* o_w      = (const float*)d_in[13];
    const float* o_b      = (const float*)d_in[14];
    float* out = (float*)d_out;

    // workspace layout (floats)
    const long long Q_ELEMS  = 4LL * 256 * 4096;  // q        : 16 MB
    const long long XS_ELEMS = 4LL * 256 * 1024;  // sampled  :  4 MB
    const long long needed   = (Q_ELEMS * 2 + XS_ELEMS * 3) * (long long)sizeof(float);
    if ((long long)ws_size < needed) return;

    float* ws   = (float*)d_ws;
    float* qbuf = ws;
    float* xs   = qbuf + Q_ELEMS;
    float* kbuf = xs + XS_ELEMS;
    float* vbuf = kbuf + XS_ELEMS;
    float* abuf = vbuf + XS_ELEMS;   // attention output, Q_ELEMS

    // 1) q projection: Bn=4, M=4096, K=256, N=256 -> 16384 wave-tiles
    {
        const long long tiles = 4LL * (4096 / 16) * (256 / 16);
        const int blocks = (int)((tiles + GEMM_WAVES - 1) / GEMM_WAVES);
        gemm_cm_bf16<<<blocks, 32 * GEMM_WAVES, 0, stream>>>(features, q_w, q_b, qbuf,
                                                             4, 4096, 256, 256);
    }
    // 2) offset branch + bilinear sampling
    offset_sample<<<64, 256, 0, stream>>>(qbuf, features, off_dw_w, off_dw_b,
                                          ln_g, ln_b, off_pw_w, xs);
    // 3) k and v projections: Bn=4, M=1024, K=256, N=256
    {
        const long long tiles = 4LL * (1024 / 16) * (256 / 16);
        const int blocks = (int)((tiles + GEMM_WAVES - 1) / GEMM_WAVES);
        gemm_cm_bf16<<<blocks, 32 * GEMM_WAVES, 0, stream>>>(xs, k_w, k_b, kbuf,
                                                             4, 1024, 256, 256);
        gemm_cm_bf16<<<blocks, 32 * GEMM_WAVES, 0, stream>>>(xs, v_w, v_b, vbuf,
                                                             4, 1024, 256, 256);
    }
    // 4) flash attention: 32 head-batches x 256 m-tiles = 8192 waves
    flash_attn_wmma<<<2048, 128, 0, stream>>>(qbuf, kbuf, vbuf, abuf);
    // 5) output projection -> d_out
    {
        const long long tiles = 4LL * (4096 / 16) * (256 / 16);
        const int blocks = (int)((tiles + GEMM_WAVES - 1) / GEMM_WAVES);
        gemm_cm_bf16<<<blocks, 32 * GEMM_WAVES, 0, stream>>>(abuf, o_w, o_b, out,
                                                             4, 4096, 256, 256);
    }
}
